// Modified_MultiheadAttention_72009421685238
// MI455X (gfx1250) — matrix operand order repaired and checked by reading, not compiled, not run on hardware
//
#include <hip/hip_runtime.h>
#include <hip/hip_bf16.h>

// CDNA5 / gfx1250: wave32, WMMA bf16 (16x16x32, f32 accumulate).
typedef __attribute__((ext_vector_type(16))) __bf16          v16bf;
typedef __attribute__((ext_vector_type(8)))  __bf16          v8bf;
typedef __attribute__((ext_vector_type(8)))  float           v8f;

#define NUM_HEADS 8
#define HD        64     // head dim
#define BQ        16     // query rows per wave
#define BS        32     // key/value rows per S-chunk (WMMA K for bf16)
#define WAVES     8      // waves per block (256 threads)
#define KPITCH    72     // LDS pitch for K chunk rows (64 + 8 pad halves)
#define VPITCH    40     // LDS pitch for V^T chunk rows (32 + 8 pad halves)

// ---------------------------------------------------------------------------
// helpers
// ---------------------------------------------------------------------------
__device__ __forceinline__ unsigned short f32_bf16(float f) {   // RNE
  unsigned u = __builtin_bit_cast(unsigned, f);
  u += 0x7FFFu + ((u >> 16) & 1u);
  return (unsigned short)(u >> 16);
}
__device__ __forceinline__ unsigned short f32_bf16_tr(float f) { // truncate (P in [0,1])
  return (unsigned short)(__builtin_bit_cast(unsigned, f) >> 16);
}

__device__ __forceinline__ v8f wmma_bf16(v16bf a, v16bf b, v8f c) {
  return __builtin_amdgcn_wmma_f32_16x16x32_bf16(false, a, false, b,
                                                 (short)0, c, false, false);
}

__device__ __forceinline__ v16bf cat8(v8bf lo, v8bf hi) {
  return __builtin_shufflevector(lo, hi, 0,1,2,3,4,5,6,7,8,9,10,11,12,13,14,15);
}

// raw v_max_num_f32 (no canonicalize ops; WMMA outputs are never sNaN)
__device__ __forceinline__ float fmax_raw(float a, float b) {
  float r;
  asm("v_max_num_f32_e32 %0, %1, %2" : "=v"(r) : "v"(a), "v"(b));
  return r;
}

// DPP-fused butterfly reductions across the 16 lanes of a half-wave:
// one VOP2+DPP instruction per step (d = op(dpp(x), x)).
__device__ __forceinline__ float row16_max(float x) {
  float r;
  asm("v_max_num_f32_dpp %0, %1, %1 quad_perm:[1,0,3,2] row_mask:0xf bank_mask:0xf"
      : "=v"(r) : "v"(x));
  asm("v_max_num_f32_dpp %0, %1, %1 quad_perm:[2,3,0,1] row_mask:0xf bank_mask:0xf"
      : "=v"(x) : "v"(r));
  asm("v_max_num_f32_dpp %0, %1, %1 row_half_mirror row_mask:0xf bank_mask:0xf"
      : "=v"(r) : "v"(x));
  asm("v_max_num_f32_dpp %0, %1, %1 row_mirror row_mask:0xf bank_mask:0xf"
      : "=v"(x) : "v"(r));
  return x;
}
__device__ __forceinline__ float row16_sum(float x) {
  float r;
  asm("v_add_f32_dpp %0, %1, %1 quad_perm:[1,0,3,2] row_mask:0xf bank_mask:0xf"
      : "=v"(r) : "v"(x));
  asm("v_add_f32_dpp %0, %1, %1 quad_perm:[2,3,0,1] row_mask:0xf bank_mask:0xf"
      : "=v"(x) : "v"(r));
  asm("v_add_f32_dpp %0, %1, %1 row_half_mirror row_mask:0xf bank_mask:0xf"
      : "=v"(r) : "v"(x));
  asm("v_add_f32_dpp %0, %1, %1 row_mirror row_mask:0xf bank_mask:0xf"
      : "=v"(x) : "v"(r));
  return x;
}

// ---------------------------------------------------------------------------
// Pre-pass: elementwise fp32 -> bf16 (memory bound; n % 4 == 0)
// ---------------------------------------------------------------------------
struct alignas(8) US4 { unsigned short s[4]; };

__global__ __launch_bounds__(256)
void cvt_f32_bf16(const float* __restrict__ src, unsigned short* __restrict__ dst,
                  long n) {
  long i = ((long)blockIdx.x * 256 + threadIdx.x) * 4;
  if (i >= n) return;
  float4 f = *(const float4*)(src + i);
  US4 o;
  o.s[0] = f32_bf16(f.x); o.s[1] = f32_bf16(f.y);
  o.s[2] = f32_bf16(f.z); o.s[3] = f32_bf16(f.w);
  *(US4*)(dst + i) = o;
}

// ---------------------------------------------------------------------------
// Pre-pass: V [b][s][h*HD+d] fp32 -> V^T bf16 [b*H+h][d][s] (LDS tile transpose)
// grid: (S/32, HD/32, B*H), block (32, 8)
// ---------------------------------------------------------------------------
__global__ __launch_bounds__(256)
void cvt_v_transpose(const float* __restrict__ v, unsigned short* __restrict__ vt,
                     int Sn, int En) {
  __shared__ unsigned short tile[32][33];
  const int bh = blockIdx.z;
  const int b = bh / NUM_HEADS, h = bh % NUM_HEADS;
  const int s0 = blockIdx.x * 32, d0 = blockIdx.y * 32;
  #pragma unroll
  for (int r = threadIdx.y; r < 32; r += 8) {
    tile[r][threadIdx.x] =
        f32_bf16(v[((long)b * Sn + s0 + r) * En + h * HD + d0 + threadIdx.x]);
  }
  __syncthreads();
  #pragma unroll
  for (int r = threadIdx.y; r < 32; r += 8) {
    vt[((long)bh * HD + d0 + r) * Sn + s0 + threadIdx.x] = tile[threadIdx.x][r];
  }
}

// ---------------------------------------------------------------------------
// Kernel: flash attention. 8 waves/block share (b, h): K/V chunks are staged
// in LDS once per block (double-buffered, software-pipelined global loads).
//   kb : bf16 K,  [b][s][h*HD+d]
//   vt : bf16 V^T [b*H+h][d][s]
//   ao : bf16 attn out [b][l][h*HD+d]
// ---------------------------------------------------------------------------
__global__ __launch_bounds__(WAVES * 32)
void attn_fwd(const float* __restrict__ q, const unsigned short* __restrict__ kb,
              const unsigned short* __restrict__ vt, unsigned short* __restrict__ ao,
              int Ln, int Sn, int En)
{
  __shared__ alignas(16) unsigned short kbuf[2][BS * KPITCH];   // 32 x 64 (+pad)
  __shared__ alignas(16) unsigned short vbuf[2][HD * VPITCH];   // 64 x 32 (+pad)
  __shared__ alignas(32) unsigned short lds_p[WAVES][BQ * BS];  // P staging

  const int tid  = threadIdx.x;
  const int wave = tid >> 5;
  const int lane = tid & 31;
  const int ln16 = lane & 15;
  const int hi   = lane >> 4;      // 0: lanes 0-15, 1: lanes 16-31
  const int c0   = hi * 8;         // unused; A fragments also use contiguous k0
  const int k0   = hi * 16;        // B-fragment contiguous K offset

  const int b  = blockIdx.z;
  const int h  = blockIdx.y;
  const int q0 = (blockIdx.x * WAVES + wave) * BQ;

  const long rowE = (long)En;
  const float*          qp = q  + ((long)b * Ln) * rowE + h * HD;
  const unsigned short* kp = kb + ((long)b * Sn) * rowE + h * HD;
  const unsigned short* vp = vt + ((long)(b * NUM_HEADS + h) * HD) * Sn;

  // cooperative staging coordinates: 16B (8 halves) per thread per matrix
  const int krow = tid >> 3, kseg = (tid & 7) * 8;   // K: 32 rows x 64 halves
  const int vrow = tid >> 2, vseg = (tid & 3) * 8;   // V: 64 rows x 32 halves
  const unsigned short* kgp = kp + (long)krow * rowE + kseg;
  const unsigned short* vgp = vp + (long)vrow * Sn + vseg;

  // ---- Q tile (16 x 64) -> two A fragments (scaled by hd^-0.5), once ----
  v16bf aq[2];
  {
    const float* qr = qp + (long)(q0 + ln16) * rowE;
    #pragma unroll
    for (int t = 0; t < 2; ++t) {
      unsigned short u[16];
      #pragma unroll
      for (int j = 0; j < 8; ++j) {
        u[j]     = f32_bf16(qr[32 * t + k0 + j]      * 0.125f);
        u[j + 8] = f32_bf16(qr[32 * t + k0 + 8 + j] * 0.125f);
      }
      aq[t] = __builtin_bit_cast(v16bf, *(US4(*)[4])u);
    }
  }

  v8f zero = {};
  v8f acc[4];
  float rmax[8], rsum[8];          // per-row state (row = r + 8*hi)
  #pragma unroll
  for (int t = 0; t < 4; ++t) acc[t] = zero;
  #pragma unroll
  for (int r = 0; r < 8; ++r) { rmax[r] = -1.0e30f; rsum[r] = 0.0f; }

  unsigned short* pbuf = lds_p[wave];

  // ---- prologue: stage chunk 0 ----
  {
    uint4 k0v = *(const uint4*)kgp;
    uint4 v0v = *(const uint4*)vgp;
    *(uint4*)&kbuf[0][krow * KPITCH + kseg] = k0v;
    *(uint4*)&vbuf[0][vrow * VPITCH + vseg] = v0v;
  }
  __syncthreads();

  int cur = 0;
  for (int sc = 0; sc < Sn; sc += BS) {
    // ---- issue next chunk's global loads (latency hides under WMMA) ----
    const bool more = (sc + BS) < Sn;
    uint4 kn, vn;
    if (more) {
      kn = *(const uint4*)(kgp + (long)(sc + BS) * rowE);
      vn = *(const uint4*)(vgp + (sc + BS));
    }

    // ---- batch-load all 4 K fragments, then the 4 score WMMAs ----
    v16bf kf[4];
    {
      const unsigned short* kr0 = &kbuf[cur][ln16 * KPITCH + k0];
      kf[0] = *(const v16bf*)(kr0);
      kf[1] = *(const v16bf*)(kr0 + 32);
      kf[2] = *(const v16bf*)(kr0 + 16 * KPITCH);
      kf[3] = *(const v16bf*)(kr0 + 16 * KPITCH + 32);
    }
    v8f s0 = zero, s1 = zero;
    s0 = wmma_bf16(aq[0], kf[0], s0);
    s0 = wmma_bf16(aq[1], kf[1], s0);
    s1 = wmma_bf16(aq[0], kf[2], s1);
    s1 = wmma_bf16(aq[1], kf[3], s1);

    // ---- online softmax: DPP-fused row reductions ----
    float p0[8], p1[8];
    #pragma unroll
    for (int r = 0; r < 8; ++r) {
      float m = row16_max(fmax_raw(s0[r], s1[r]));
      float newm = fmax_raw(rmax[r], m);
      float corr = __expf(rmax[r] - newm);
      rmax[r] = newm;
      float e0 = __expf(s0[r] - newm);
      float e1 = __expf(s1[r] - newm);
      float t = row16_sum(e0 + e1);
      rsum[r] = rsum[r] * corr + t;
      acc[0][r] *= corr; acc[1][r] *= corr;
      acc[2][r] *= corr; acc[3][r] *= corr;
      p0[r] = e0; p1[r] = e1;
    }

    // ---- batch-load V fragments; stage P through LDS (C -> A layout) ----
    v16bf vf[4];
    #pragma unroll
    for (int dt = 0; dt < 4; ++dt)
      vf[dt] = *(const v16bf*)(&vbuf[cur][(dt * 16 + ln16) * VPITCH + k0]);

    #pragma unroll
    for (int r = 0; r < 8; ++r) {
      int m = r + hi * 8;
      pbuf[m * BS + ln16]      = f32_bf16_tr(p0[r]);
      pbuf[m * BS + 16 + ln16] = f32_bf16_tr(p1[r]);
    }
    asm volatile("" ::: "memory");   // keep same-wave LDS store->load order
    const unsigned short* pr = &pbuf[ln16 * BS];
    v16bf pa = cat8(*(const v8bf*)(pr + k0), *(const v8bf*)(pr + k0 + 8));

    // ---- acc += P(16x32) . V(32x64) ----
    #pragma unroll
    for (int dt = 0; dt < 4; ++dt)
      acc[dt] = wmma_bf16(pa, vf[dt], acc[dt]);

    // ---- commit next chunk to the other buffer ----
    if (more) {
      *(uint4*)&kbuf[cur ^ 1][krow * KPITCH + kseg] = kn;
      *(uint4*)&vbuf[cur ^ 1][vrow * VPITCH + vseg] = vn;
    }
    __syncthreads();
    cur ^= 1;
  }

  // ---- normalize, store bf16 attn (feeds projection GEMM directly) ----
  unsigned short* op = ao + ((long)b * Ln + q0) * rowE + h * HD;
  #pragma unroll
  for (int r = 0; r < 8; ++r) {
    float inv = 1.0f / rsum[r];
    #pragma unroll
    for (int dt = 0; dt < 4; ++dt)
      op[(long)(r + hi * 8) * rowE + dt * 16 + ln16] = f32_bf16(acc[dt][r] * inv);
  }
}

// ---------------------------------------------------------------------------
// Kernel: out_z[m,n] = sum_k attn_bf16[m,k] * W_z_bf16[n,k] + bias_z[n]
// 8 waves arranged 4x2 -> block covers 64 rows x 128 cols.
// ---------------------------------------------------------------------------
__global__ __launch_bounds__(WAVES * 32)
void proj_gemm(const unsigned short* __restrict__ A,
               const unsigned short* __restrict__ W1, const float* __restrict__ B1,
               const unsigned short* __restrict__ W2, const float* __restrict__ B2,
               float* __restrict__ out, int M, int En)
{
  const int wave = threadIdx.x >> 5;
  const int lane = threadIdx.x & 31;
  const int ln16 = lane & 15;
  const int hi   = lane >> 4;
  const int c0   = hi * 8;
  const int k0   = hi * 16;

  const unsigned short* W  = blockIdx.z ? W2 : W1;
  const float*          Bi = blockIdx.z ? B2 : B1;
  float* op = out + (long)blockIdx.z * (long)M * En;

  const int m0 = blockIdx.x * 64  + (wave & 3) * 16;   // 4 m-tiles
  const int n0 = blockIdx.y * 128 + (wave >> 2) * 64;  // 2 n-tiles

  v8f zero = {};
  v8f acc[4];
  #pragma unroll
  for (int t = 0; t < 4; ++t) acc[t] = zero;

  const unsigned short* ar = A + (long)(m0 + ln16) * En;
  for (int kk = 0; kk < En; kk += 32) {
    v16bf af = cat8(*(const v8bf*)(ar + kk + k0),
                    *(const v8bf*)(ar + kk + k0 + 8));
    #pragma unroll
    for (int nt = 0; nt < 4; ++nt) {
      const unsigned short* wr = W + (long)(n0 + nt * 16 + ln16) * En + kk + k0;
      acc[nt] = wmma_bf16(af, *(const v16bf*)wr, acc[nt]);
    }
  }

  #pragma unroll
  for (int nt = 0; nt < 4; ++nt) {
    float bias = Bi[n0 + nt * 16 + ln16];
    #pragma unroll
    for (int r = 0; r < 8; ++r)
      op[(long)(m0 + r + hi * 8) * En + n0 + nt * 16 + ln16] = acc[nt][r] + bias;
  }
}

// ---------------------------------------------------------------------------
extern "C" void kernel_launch(void* const* d_in, const int* in_sizes, int n_in,
                              void* d_out, int out_size, void* d_ws, size_t ws_size,
                              hipStream_t stream) {
  (void)in_sizes; (void)n_in; (void)out_size; (void)ws_size;

  const float* q  = (const float*)d_in[0];
  const float* k  = (const float*)d_in[1];
  const float* v  = (const float*)d_in[2];
  const float* w1 = (const float*)d_in[3];
  const float* b1 = (const float*)d_in[4];
  const float* w2 = (const float*)d_in[5];
  const float* b2 = (const float*)d_in[6];
  float* out = (float*)d_out;

  const int B = 16, L = 1024, S = 1024, E = 512;
  const long BSE = (long)B * S * E;       // 8,388,608
  const long EE  = (long)E * E;           // 262,144

  // workspace layout (bf16): K | V^T | attn | W1 | W2  (~51.4 MB)
  unsigned short* ws  = (unsigned short*)d_ws;
  unsigned short* kb  = ws;
  unsigned short* vtb = kb  + BSE;
  unsigned short* ab  = vtb + BSE;
  unsigned short* w1b = ab  + BSE;
  unsigned short* w2b = w1b + EE;

  // pre-passes: bulk fp32->bf16 (memory bound)
  cvt_f32_bf16<<<dim3((unsigned)(BSE / 1024)), 256, 0, stream>>>(k,  kb,  BSE);
  cvt_f32_bf16<<<dim3((unsigned)(EE  / 1024)), 256, 0, stream>>>(w1, w1b, EE);
  cvt_f32_bf16<<<dim3((unsigned)(EE  / 1024)), 256, 0, stream>>>(w2, w2b, EE);
  cvt_v_transpose<<<dim3(S / 32, HD / 32, B * NUM_HEADS), dim3(32, 8), 0, stream>>>(
      v, vtb, S, E);

  // attention: 8 waves/block * 16 rows = 128 q-rows per block
  dim3 gridA(L / (WAVES * BQ), NUM_HEADS, B);        // (8, 8, 16)
  attn_fwd<<<gridA, WAVES * 32, 0, stream>>>(q, kb, vtb, ab, L, S, E);

  // dual projection: 64 rows x 128 cols per block, z = weight index
  dim3 gridG((B * L) / 64, E / 128, 2);              // (256, 4, 2)
  proj_gemm<<<gridG, WAVES * 32, 0, stream>>>(ab, w1b, b1, w2b, b2, out, B * L, E);
}
